// F2FConv3d_54640573939773
// MI455X (gfx1250) — matrix operand order repaired and checked by reading, not compiled, not run on hardware
//
#include <hip/hip_runtime.h>

typedef __bf16 bf16;
typedef __attribute__((ext_vector_type(16))) __bf16 v16bf;
typedef __attribute__((ext_vector_type(8)))  float  v8f;

#define F_FACETS 400000
#define TEX_PER  4
#define T_TEX    (F_FACETS * TEX_PER)
#define NB       9
#define NTILES   (T_TEX / 16)      // 100000 tiles of 16 texture rows
#define BN_EPS   1e-3f
#define XPITCH   68                // floats per LDS row: 68*4 = 272 B = 17*16 (b128-aligned)

// ---- CDNA5 async global->LDS copy (ASYNCcnt-tracked) ----------------------
__device__ __forceinline__ void async_b128(unsigned int lds_byte_off, const float* g)
{
    asm volatile("global_load_async_to_lds_b128 %0, %1, off"
                 :: "v"(lds_byte_off),
                    "v"((unsigned long long)(uintptr_t)g)
                 : "memory");
}
__device__ __forceinline__ unsigned int lds_off(const void* p)
{
    return (unsigned int)(uintptr_t)p;   // low 32 bits of flat addr = LDS offset
}

// ---------------------------------------------------------------------------
// Main kernel: per 16-texture tile, compute contrib[16,64] via bf16 WMMA,
// reduce groups of 4 rows -> 4 facets, apply /cnt + bias + ReLU, store,
// and accumulate per-channel BN partial sums.
// Block = 128 threads = 4 waves; wave w owns output channels [16w, 16w+16).
// x/bary tiles are double-buffered in LDS and prefetched with
// global_load_async_to_lds_b128, gated by s_wait_asynccnt.
// ---------------------------------------------------------------------------
__global__ __launch_bounds__(128, 1) void f2f_main(
    const float* __restrict__ x,      // [T_TEX, 64]
    const float* __restrict__ bary,   // [T_TEX, 9]
    const float* __restrict__ W,      // [64, 64, 9]  (O, I, K)
    const float* __restrict__ bias,   // [64]
    const int*   __restrict__ ntex,   // [F_FACETS]
    float*       __restrict__ out,    // [F_FACETS, 64]
    float*       __restrict__ bn_acc) // [0:64] sum, [64:128] sumsq
{
    __shared__ __align__(16) float xs[2][16 * XPITCH];
    __shared__ __align__(16) float bs[2][16 * NB];

    const int tid  = threadIdx.x;
    const int lane = tid & 31;
    const int wave = tid >> 5;
    const int l15  = lane & 15;
    const int hi   = lane >> 4;
    const int n    = wave * 16 + l15;           // this lane's output channel

    // per-thread async-copy assignments: 8 consecutive floats of the 16x64
    // x tile (2 x b128, same LDS row), plus bary (36 threads x b128).
    const int e0   = tid * 8;
    const int xrow = e0 >> 6, xcol = e0 & 63;
    const unsigned int xdst0 = lds_off(&xs[0][xrow * XPITCH + xcol]);
    const unsigned int xdst1 = lds_off(&xs[1][xrow * XPITCH + xcol]);
    const unsigned int bdst0 = lds_off(&bs[0][4 * tid]);
    const unsigned int bdst1 = lds_off(&bs[1][4 * tid]);

    auto issue_tile = [&](int t, int buf) {
        const float* xg = x + (size_t)t * 1024 + e0;
        const unsigned int xd = buf ? xdst1 : xdst0;
        async_b128(xd,      xg);
        async_b128(xd + 16, xg + 4);
        if (tid < 36)
            async_b128(buf ? bdst1 : bdst0, bary + (size_t)t * 144 + 4 * tid);
    };

    const int stride = (int)gridDim.x;
    int tile = blockIdx.x;
    if (tile < NTILES) issue_tile(tile, 0);     // prefetch first tile early

    // ---- Preload B fragments into registers: bfrag[chunk][k] -------------
    // GEMM_k: contrib_k[m,o] = sum_i x[m,i] * W[o,i,k]; B_k[i,o] = W[o,i,k].
    // 32x16 bf16 B layout: lane group hi: elems 0..7 -> K = 8*hi + e,
    // elems 8..15 -> K = 16 + 8*hi + (e-8); column = l15.
    v16bf bfrag[2][NB];
#pragma unroll
    for (int c = 0; c < 2; ++c) {
#pragma unroll
        for (int k = 0; k < NB; ++k) {
#pragma unroll
            for (int j = 0; j < 8; ++j) {
                const int i0 = c * 32 + hi * 8 + 2 * j + (j >= 4 ? 8 : 0);
                bfrag[c][k][2 * j]     = (bf16)W[(n * 64 + i0) * 9 + k];
                bfrag[c][k][2 * j + 1] = (bf16)W[(n * 64 + i0 + 1) * 9 + k];
            }
        }
    }
    const float bn = bias[n];

    float bnS = 0.f, bnQ = 0.f;
    int buf = 0;

    for (; tile < NTILES; tile += stride, buf ^= 1) {
        // prefetch next tile into the other buffer, then wait for current.
        // Completion is in-order; the next batch has >=2 ops per wave, so
        // asynccnt<=2 guarantees the current batch has fully landed.
        const int nxt = tile + stride;
        if (nxt < NTILES) {
            issue_tile(nxt, buf ^ 1);
            asm volatile("s_wait_asynccnt 2" ::: "memory");
        } else {
            asm volatile("s_wait_asynccnt 0" ::: "memory");
        }
        __syncthreads();                       // current tile visible to all waves

        // ---- build A fragments (16x32 bf16, two i-chunks), unscaled ------
        // A layout: lane l15 = row m; elems 0..7 -> K = 8*hi + e,
        //           elems 8..15 -> K = 16 + 8*hi + (e-8).
        const float* xr = &xs[buf][l15 * XPITCH];
        v16bf afrag[2];
#pragma unroll
        for (int c = 0; c < 2; ++c) {
#pragma unroll
            for (int e = 0; e < 8; ++e) {
                afrag[c][e]     = (bf16)xr[c * 32 + hi * 8 + e];
                afrag[c][8 + e] = (bf16)xr[c * 32 + 16 + hi * 8 + e];
            }
        }

        // ---- 9 x (2-chunk WMMA) with bary-weighted f32 epilogue ----------
        const float* br = &bs[buf][8 * hi * NB];
        float contrib[8] = {0.f, 0.f, 0.f, 0.f, 0.f, 0.f, 0.f, 0.f};
#pragma unroll
        for (int k = 0; k < NB; ++k) {
            v8f acc = {};
            acc = __builtin_amdgcn_wmma_f32_16x16x32_bf16(
                false, afrag[0], false, bfrag[0][k], (short)0, acc, false, false);
            acc = __builtin_amdgcn_wmma_f32_16x16x32_bf16(
                false, afrag[1], false, bfrag[1][k], (short)0, acc, false, false);
            // C/D layout: VGPR r holds row m = r + 8*hi for this lane group
#pragma unroll
            for (int r = 0; r < 8; ++r)
                contrib[r] += acc[r] * br[r * NB + k];   // same-addr broadcast read
        }

        // ---- facet reduction (4 consecutive texture rows per facet) ------
        const float fs0 = contrib[0] + contrib[1] + contrib[2] + contrib[3];
        const float fs1 = contrib[4] + contrib[5] + contrib[6] + contrib[7];
        const int f0 = 4 * tile + 2 * hi;    // rows 0-3 (hi=0) / 8-11 (hi=1)
        const int f1 = f0 + 1;               // rows 4-7 / 12-15

        float o0 = fs0 / (float)ntex[f0] + bn;
        float o1 = fs1 / (float)ntex[f1] + bn;
        o0 = fmaxf(o0, 0.f);
        o1 = fmaxf(o1, 0.f);

        out[(size_t)f0 * 64 + n] = o0;
        out[(size_t)f1 * 64 + n] = o1;

        bnS += o0 + o1;
        bnQ += o0 * o0 + o1 * o1;

        __syncthreads();   // all waves done reading buf before it is refilled
    }

    atomicAdd(&bn_acc[n], bnS);
    atomicAdd(&bn_acc[64 + n], bnQ);
}

// ---------------------------------------------------------------------------
__global__ void f2f_init(float* ws)
{
    ws[threadIdx.x] = 0.f;            // zero the 128 BN accumulators
}

__global__ void f2f_stats(const float* __restrict__ acc,
                          const float* __restrict__ gamma,
                          const float* __restrict__ beta,
                          float* __restrict__ sc_sh)   // [0:64] scale, [64:128] shift
{
    const int n = threadIdx.x;
    const float inv = 1.0f / (float)F_FACETS;
    const float mu  = acc[n] * inv;
    const float var = acc[64 + n] * inv - mu * mu;
    const float sc  = gamma[n] * rsqrtf(var + BN_EPS);
    sc_sh[n]      = sc;
    sc_sh[64 + n] = beta[n] - mu * sc;
}

__global__ __launch_bounds__(256) void f2f_norm(
    float* __restrict__ out,
    const float* __restrict__ sc_sh, int total)
{
    const int e = blockIdx.x * 256 + threadIdx.x;
    if (e < total) {
        const int n = e & 63;
        out[e] = out[e] * sc_sh[n] + sc_sh[64 + n];
    }
}

// ---------------------------------------------------------------------------
extern "C" void kernel_launch(void* const* d_in, const int* in_sizes, int n_in,
                              void* d_out, int out_size, void* d_ws, size_t ws_size,
                              hipStream_t stream)
{
    const float* x     = (const float*)d_in[0];   // input_texture [T,64]
    const float* bary  = (const float*)d_in[1];   // bary_coeff   [T,9]
    const float* W     = (const float*)d_in[2];   // weights      [64,64,9]
    const float* bias  = (const float*)d_in[3];   // biases       [1,64]
    const float* gamma = (const float*)d_in[4];   // gamma        [64]
    const float* beta  = (const float*)d_in[5];   // beta         [64]
    const int*   ntex  = (const int*)d_in[6];     // num_texture  [F]
    float*       out   = (float*)d_out;           // [F,64]
    float*       ws    = (float*)d_ws;            // >= 256 floats

    f2f_init<<<1, 128, 0, stream>>>(ws);
    f2f_main<<<2048, 128, 0, stream>>>(x, bary, W, bias, ntex, out, ws);
    f2f_stats<<<1, 64, 0, stream>>>(ws, gamma, beta, ws + 128);
    const int total = F_FACETS * 64;
    f2f_norm<<<(total + 255) / 256, 256, 0, stream>>>(out, ws + 128, total);
}
